// GATLayer_5059471475380
// MI455X (gfx1250) — matrix operand order repaired and checked by reading, not compiled, not run on hardware
//
#include <hip/hip_runtime.h>
#include <hip/hip_bf16.h>
#include <math.h>

typedef __attribute__((ext_vector_type(2)))  float  v2f;
typedef __attribute__((ext_vector_type(8)))  float  v8f;
typedef __attribute__((ext_vector_type(8)))  __bf16 v8bf;
typedef __attribute__((ext_vector_type(16))) __bf16 v16bf;

#define ALPHA 0.2f
#define NEG_BIG (-9.0e15f)

static constexpr int B = 16, N = 1024, F = 256;

// ---------------------------------------------------------------------------
// Kernel 1: h = x @ W   (f32 WMMA 16x16x4, full f32 precision for the logits)
// Block = 256 threads = 8 waves; each wave owns a 16-row x 256-col tile.
// A frag (16x4 f32): lane l holds x[row=l%16][k + 2*(l/16) + {0,1}]  (v2f)
// B frag (4x16 f32): lane l holds W[k + 2*(l/16) + {0,1}][f0 + l%16]
// C (16x16 f32):     vgpr r, lane l -> row r + 8*(l/16), col f0 + l%16
// ---------------------------------------------------------------------------
__global__ __launch_bounds__(256) void gat_proj(const float* __restrict__ x,
                                                const float* __restrict__ W,
                                                float* __restrict__ h) {
  const int lane = threadIdx.x & 31;
  const int wave = threadIdx.x >> 5;
  const int batch = blockIdx.y;
  const int r0 = blockIdx.x * 128 + wave * 16;
  const int lh = lane >> 4;   // lane half: 0/1
  const int ll = lane & 15;

  v8f acc[16];
  #pragma unroll
  for (int t = 0; t < 16; ++t) acc[t] = v8f{};

  const float* xp = x + ((size_t)batch * N + r0 + ll) * F + 2 * lh;
  const float* wp = W + (size_t)(2 * lh) * F + ll;

  for (int k = 0; k < F; k += 4) {
    v2f a = *(const v2f*)(xp + k);          // K = k+2*lh, k+2*lh+1
    const float* wk = wp + (size_t)k * F;
    #pragma unroll
    for (int t = 0; t < 16; ++t) {
      v2f bf;
      bf.x = wk[t * 16];                    // W[k+2lh  ][t*16+ll]
      bf.y = wk[t * 16 + F];                // W[k+2lh+1][t*16+ll]
      acc[t] = __builtin_amdgcn_wmma_f32_16x16x4_f32(
          false, a, false, bf, (short)0, acc[t], false, false);
    }
  }

  float* hp = h + ((size_t)batch * N + r0 + 8 * lh) * F + ll;
  #pragma unroll
  for (int t = 0; t < 16; ++t)
    #pragma unroll
    for (int r = 0; r < 8; ++r)
      hp[(size_t)r * F + t * 16] = acc[t][r];
}

// ---------------------------------------------------------------------------
// Kernel 2: el = h @ a_l, er = h @ a_r  (tiny; one wave per node, shfl reduce)
// ---------------------------------------------------------------------------
__global__ __launch_bounds__(256) void gat_el_er(const float* __restrict__ h,
                                                 const float* __restrict__ a,
                                                 float* __restrict__ el,
                                                 float* __restrict__ er) {
  const int lane = threadIdx.x & 31;
  const int wave = threadIdx.x >> 5;
  const int node = blockIdx.x * 8 + wave;   // 0 .. B*N-1
  const float* hp = h + (size_t)node * F;
  float sl = 0.f, sr = 0.f;
  #pragma unroll
  for (int k = lane; k < F; k += 32) {
    float v = hp[k];
    sl += v * a[k];
    sr += v * a[F + k];
  }
  #pragma unroll
  for (int off = 16; off > 0; off >>= 1) {
    sl += __shfl_down(sl, off, 32);
    sr += __shfl_down(sr, off, 32);
  }
  if (lane == 0) { el[node] = sl; er[node] = sr; }
}

// ---------------------------------------------------------------------------
// Kernel 3: masked-softmax attention + aggregation, flash-style.
// Block = 256 threads = 8 waves handles one (batch, 16-row) tile.
//  Pass 1: two sweeps over j give per-row max and sum(exp) (attn never hits HBM)
//  Pass 2: per 64-j block, build P (bf16) + transposed h tile (bf16) in LDS,
//          then bf16 WMMA 16x16x32 with f32 accumulate. Wave w owns f cols
//          [32w, 32w+32) = two 16-wide C tiles.
// bf16 A frag: lane l = row l%16; K chunks [8lh,8lh+8) and [16+8lh,24+8lh)
// bf16 B frag: lane l = col l%16; K chunks [8lh,8lh+8) and [16+8lh,24+8lh)
// ---------------------------------------------------------------------------
__global__ __launch_bounds__(256) void gat_attn(const float* __restrict__ h,
                                                const float* __restrict__ el,
                                                const float* __restrict__ er,
                                                const int* __restrict__ adj,
                                                float* __restrict__ out) {
  __shared__ float  s_el[16];
  __shared__ float  s_m[16];
  __shared__ float  s_s[16];
  __shared__ float  s_red[16][16];
  __shared__ __bf16 s_P[16][72];    // 16 rows x 64 j, padded (144B rows)
  __shared__ __bf16 s_hT[256][72];  // 256 f x 64 j (transposed h), padded

  const int t = threadIdx.x;
  const int lane = t & 31;
  const int wave = t >> 5;
  const int batch = blockIdx.y;
  const int i0 = blockIdx.x * 16;
  const int lh = lane >> 4, ll = lane & 15;

  const float* erp = er + (size_t)batch * N;
  const int* adjb = adj + ((size_t)batch * N + i0) * N;
  const float* hb = h + (size_t)batch * N * F;

  if (t < 16) s_el[t] = el[(size_t)batch * N + i0 + t];
  __syncthreads();

  const int r = t >> 4;                 // row 0..15 (16 threads per row)
  const int c = t & 15;
  const int* adjr = adjb + (size_t)r * N;
  const float eli = s_el[r];

  // pass 1a: row max of masked leaky-relu logits
  float mx = NEG_BIG;
  for (int j = c; j < N; j += 16) {
    float e = eli + erp[j];
    e = e > 0.f ? e : ALPHA * e;
    e = adjr[j] > 0 ? e : NEG_BIG;
    mx = fmaxf(mx, e);
  }
  s_red[r][c] = mx;
  __syncthreads();
  if (t < 16) {
    float m = s_red[t][0];
    #pragma unroll
    for (int q = 1; q < 16; ++q) m = fmaxf(m, s_red[t][q]);
    s_m[t] = m;
  }
  __syncthreads();

  // pass 1b: row sum of exp(e - max)
  const float mrow = s_m[r];
  float sum = 0.f;
  for (int j = c; j < N; j += 16) {
    float e = eli + erp[j];
    e = e > 0.f ? e : ALPHA * e;
    e = adjr[j] > 0 ? e : NEG_BIG;
    sum += expf(e - mrow);
  }
  s_red[r][c] = sum;
  __syncthreads();
  if (t < 16) {
    float sm = 0.f;
    #pragma unroll
    for (int q = 0; q < 16; ++q) sm += s_red[t][q];
    s_s[t] = sm;
  }
  __syncthreads();
  const float inv_s = 1.0f / s_s[r];

  // pass 2: out[i,:] = sum_j P[i,j] * h[j,:]
  v8f acc0 = v8f{}, acc1 = v8f{};

  for (int jb = 0; jb < N; jb += 64) {
    __builtin_prefetch(adjr + jb + 64, 0, 0);
    // stage transposed bf16 h tile: s_hT[f][jj] = h[jb+jj][f]
    #pragma unroll 8
    for (int jj = 0; jj < 64; ++jj)
      s_hT[t][jj] = (__bf16)hb[(size_t)(jb + jj) * F + t];
    // build normalized P tile in bf16: 4 values per thread
    #pragma unroll
    for (int q = 0; q < 4; ++q) {
      int j = jb + c + 16 * q;
      float e = eli + erp[j];
      e = e > 0.f ? e : ALPHA * e;
      e = adjr[j] > 0 ? e : NEG_BIG;
      s_P[r][c + 16 * q] = (__bf16)(expf(e - mrow) * inv_s);
    }
    __syncthreads();

    #pragma unroll
    for (int k0 = 0; k0 < 64; k0 += 32) {
      // A fragment from s_P
      v8bf pa0 = *(const v8bf*)&s_P[ll][k0 + 8 * lh];
      v8bf pa1 = *(const v8bf*)&s_P[ll][k0 + 16 + 8 * lh];
      v16bf af;
      #pragma unroll
      for (int i = 0; i < 8; ++i) { af[i] = pa0[i]; af[i + 8] = pa1[i]; }
      // B fragments from s_hT (two 16-wide f tiles per wave)
      const __bf16* hc0 = &s_hT[wave * 32 + ll][k0 + 8 * lh];
      const __bf16* hc1 = hc0 + 16 * 72;
      v8bf b0a = *(const v8bf*)hc0, b0b = *(const v8bf*)(hc0 + 16);
      v8bf b1a = *(const v8bf*)hc1, b1b = *(const v8bf*)(hc1 + 16);
      v16bf bf0, bf1;
      #pragma unroll
      for (int i = 0; i < 8; ++i) {
        bf0[i] = b0a[i]; bf0[i + 8] = b0b[i];
        bf1[i] = b1a[i]; bf1[i + 8] = b1b[i];
      }
      acc0 = __builtin_amdgcn_wmma_f32_16x16x32_bf16(
          false, af, false, bf0, (short)0, acc0, false, false);
      acc1 = __builtin_amdgcn_wmma_f32_16x16x32_bf16(
          false, af, false, bf1, (short)0, acc1, false, false);
    }
    __syncthreads();
  }

  float* op = out + ((size_t)batch * N + i0 + 8 * lh) * F + wave * 32 + ll;
  #pragma unroll
  for (int rr = 0; rr < 8; ++rr) {
    op[(size_t)rr * F] = acc0[rr];
    op[(size_t)rr * F + 16] = acc1[rr];
  }
}

// ---------------------------------------------------------------------------
extern "C" void kernel_launch(void* const* d_in, const int* in_sizes, int n_in,
                              void* d_out, int out_size, void* d_ws, size_t ws_size,
                              hipStream_t stream) {
  const float* x   = (const float*)d_in[0];   // (B, N, F)
  const int*   adj = (const int*)d_in[1];     // (B, N, N)
  const float* W   = (const float*)d_in[2];   // (F, F)
  const float* a   = (const float*)d_in[3];   // (2F, 1)
  float* out = (float*)d_out;                 // (B, N, F)

  float* h  = (float*)d_ws;                   // B*N*F floats (~16.8 MB)
  float* el = h + (size_t)B * N * F;          // B*N floats
  float* er = el + (size_t)B * N;             // B*N floats

  gat_proj <<<dim3(N / 128, B), 256, 0, stream>>>(x, W, h);
  gat_el_er<<<dim3((B * N) / 8), 256, 0, stream>>>(h, a, el, er);
  gat_attn <<<dim3(N / 16, B), 256, 0, stream>>>(h, el, er, adj, out);
}
